// Net_6408091205738
// MI455X (gfx1250) — compile-verified, matrix operand order checked by reading, not run on hardware
//
#include <hip/hip_runtime.h>

// RGCN 2-layer forward for MI455X (gfx1250).
// Transform-then-scatter: per relation r, y = feat @ w_r via bf16x3 WMMA GEMM
// (fp32-accurate: hi*hi + hi*lo + lo*hi with f32 accumulation), then atomic
// scatter y[src]*inv_cnt[dst,r] into out[dst]. The 25.6MB atomic target and y
// stay L2-resident (192MB L2). Feature matrix is pre-split to bf16 hi/lo once
// per layer so the GEMM inner loop is pure 16B loads + v_wmma.
// Workspace need: ~80 MB.

#define N_NODES 50000
#define DIM     128
#define NEDGE   600000
#define NREL    8

typedef __attribute__((ext_vector_type(16))) __bf16 v16bf;
typedef __attribute__((ext_vector_type(8)))  float  v8f;

__device__ __forceinline__ unsigned short f2bf(float f) {
    union { float f; unsigned u; } v; v.f = f;
    unsigned u = v.u;
    unsigned r = u + 0x7FFFu + ((u >> 16) & 1u);   // round-to-nearest-even
    return (unsigned short)(r >> 16);
}
__device__ __forceinline__ float bf2f(unsigned short h) {
    union { unsigned u; float f; } v; v.u = ((unsigned)h) << 16;
    return v.f;
}

__global__ void zero_f(float* p, int n) {
    int i = blockIdx.x * blockDim.x + threadIdx.x;
    if (i < n) p[i] = 0.0f;
}

__global__ void count_edges(const int* __restrict__ ei, const int* __restrict__ et,
                            float* __restrict__ cnt) {
    int e = blockIdx.x * blockDim.x + threadIdx.x;
    if (e >= NEDGE) return;
    int dst = ei[NEDGE + e];
    int r   = et[e];
    unsafeAtomicAdd(&cnt[dst * NREL + r], 1.0f);
}

__global__ void invert_cnt(float* p, int n) {
    int i = blockIdx.x * blockDim.x + threadIdx.x;
    if (i < n) p[i] = 1.0f / fmaxf(p[i], 1.0f);
}

// Pre-transpose + bf16 hi/lo split of weights.
// Output layout: wt[j][e][d] = (j<8 ? w[j][d][e] : root[d][e]),  j=0..8
__global__ void split_weights(const float* __restrict__ w, const float* __restrict__ root,
                              unsigned short* __restrict__ hi, unsigned short* __restrict__ lo) {
    int i = blockIdx.x * blockDim.x + threadIdx.x;
    if (i >= 9 * DIM * DIM) return;
    int j   = i / (DIM * DIM);
    int rem = i % (DIM * DIM);
    int e = rem / DIM;      // output feature (N dim of GEMM)
    int d = rem % DIM;      // input feature  (K dim of GEMM)
    float val = (j < NREL) ? w[(size_t)j * DIM * DIM + d * DIM + e]
                           : root[d * DIM + e];
    unsigned short h = f2bf(val);
    hi[i] = h;
    lo[i] = f2bf(val - bf2f(h));
}

// bf16 hi/lo split of the feature matrix (row-major [N][D]), 4 elems/thread.
__global__ void split_feat(const float* __restrict__ feat,
                           unsigned short* __restrict__ fhi,
                           unsigned short* __restrict__ flo) {
    int i = blockIdx.x * blockDim.x + threadIdx.x;       // over N*D/4
    if (i >= N_NODES * DIM / 4) return;
    float4 v = ((const float4*)feat)[i];
    float a[4] = { v.x, v.y, v.z, v.w };
    ushort4 hh, ll;
    unsigned short* hp = (unsigned short*)&hh;
    unsigned short* lp = (unsigned short*)&ll;
#pragma unroll
    for (int k = 0; k < 4; ++k) {
        unsigned short h = f2bf(a[k]);
        hp[k] = h;
        lp[k] = f2bf(a[k] - bf2f(h));
    }
    ((ushort4*)fhi)[i] = hh;
    ((ushort4*)flo)[i] = ll;
}

// C[m][e] = sum_d A[m][d] * Wt[e][d]  (+ optional bias[e])
// One wave per 16x16 output tile; K=128 in four K=32 steps; bf16x3 emulation.
// A and B are both pre-split bf16 (hi/lo) -> inner loop is 16B loads + WMMA.
__global__ __launch_bounds__(256) void gemm_bf16x3(
    const unsigned short* __restrict__ Ahi,   // [N][128] bf16 bits
    const unsigned short* __restrict__ Alo,
    const unsigned short* __restrict__ Bhi,   // [128][128] bf16 bits, (e, d) row-major
    const unsigned short* __restrict__ Blo,
    const float* __restrict__ bias,           // nullptr or [128]
    float* __restrict__ C)
{
    int wid   = blockIdx.x * (blockDim.x >> 5) + (threadIdx.x >> 5);
    int lane  = threadIdx.x & 31;
    int mtile = wid >> 3;          // 3125 m-tiles (exact: 50000/16)
    int ntile = wid & 7;           // 8 n-tiles (128/16)
    int half  = lane >> 4;
    int l16   = lane & 15;
    int kbase = half * 8;
    int ncol  = ntile * 16 + l16;

    const unsigned short* ahrow = Ahi + (size_t)(mtile * 16 + l16) * DIM;
    const unsigned short* alrow = Alo + (size_t)(mtile * 16 + l16) * DIM;
    const unsigned short* bhrow = Bhi + (size_t)ncol * DIM;
    const unsigned short* blrow = Blo + (size_t)ncol * DIM;

    v8f acc = {};
#pragma unroll
    for (int kb = 0; kb < DIM; kb += 32) {
        union { v16bf v; unsigned short s[16]; } Ah, Al, Bh, Bl;
        // lane holds K = {kb+kbase .. +7} and {kb+16+kbase .. +7}: two uint4 loads each
        const uint4* ah = (const uint4*)(ahrow + kb + kbase);
        const uint4* al = (const uint4*)(alrow + kb + kbase);
        const uint4* bh = (const uint4*)(bhrow + kb + kbase);
        const uint4* bl = (const uint4*)(blrow + kb + kbase);
        *(uint4*)&Ah.s[0] = ah[0];  *(uint4*)&Ah.s[8] = ah[2];
        *(uint4*)&Al.s[0] = al[0];  *(uint4*)&Al.s[8] = al[2];
        *(uint4*)&Bh.s[0] = bh[0];  *(uint4*)&Bh.s[8] = bh[2];
        *(uint4*)&Bl.s[0] = bl[0];  *(uint4*)&Bl.s[8] = bl[2];

        // bf16x3: hi*hi + hi*lo + lo*hi (fp32 accumulate)
        acc = __builtin_amdgcn_wmma_f32_16x16x32_bf16(false, Ah.v, false, Bh.v,
                                                      (short)0, acc, false, false);
        acc = __builtin_amdgcn_wmma_f32_16x16x32_bf16(false, Ah.v, false, Bl.v,
                                                      (short)0, acc, false, false);
        acc = __builtin_amdgcn_wmma_f32_16x16x32_bf16(false, Al.v, false, Bh.v,
                                                      (short)0, acc, false, false);
    }

    float bb = bias ? bias[ncol] : 0.0f;
    float* cp = C + (size_t)(mtile * 16 + half * 8) * DIM + ncol;
#pragma unroll
    for (int v = 0; v < 8; ++v) cp[(size_t)v * DIM] = acc[v] + bb;
}

// out[dst] += y[src] * inv[dst*NREL+rel] for edges of this relation.
// 32 lanes per edge, float4 per lane -> 128 floats.
__global__ void scatter_rel(const int* __restrict__ ei, const int* __restrict__ et,
                            const float* __restrict__ y, const float* __restrict__ inv,
                            float* __restrict__ out, int rel)
{
    int t = blockIdx.x * blockDim.x + threadIdx.x;
    int e = t >> 5;
    if (e >= NEDGE) return;
    if (et[e] != rel) return;
    int c   = (t & 31) << 2;
    int src = ei[e];
    int dst = ei[NEDGE + e];
    float s = inv[dst * NREL + rel];
    float4 v = *(const float4*)(y + (size_t)src * DIM + c);
    float* o = out + (size_t)dst * DIM + c;
    unsafeAtomicAdd(o + 0, v.x * s);
    unsafeAtomicAdd(o + 1, v.y * s);
    unsafeAtomicAdd(o + 2, v.z * s);
    unsafeAtomicAdd(o + 3, v.w * s);
}

__global__ void relu_k(float* p, int n4) {
    int i = blockIdx.x * blockDim.x + threadIdx.x;
    if (i >= n4) return;
    float4* p4 = (float4*)p;
    float4 v = p4[i];
    v.x = fmaxf(v.x, 0.0f); v.y = fmaxf(v.y, 0.0f);
    v.z = fmaxf(v.z, 0.0f); v.w = fmaxf(v.w, 0.0f);
    p4[i] = v;
}

extern "C" void kernel_launch(void* const* d_in, const int* in_sizes, int n_in,
                              void* d_out, int out_size, void* d_ws, size_t ws_size,
                              hipStream_t stream) {
    const float* x     = (const float*)d_in[0];
    const int*   ei    = (const int*)  d_in[1];   // [2][E]: row0=src, row1=dst
    const int*   et    = (const int*)  d_in[2];   // [E]
    const float* w1    = (const float*)d_in[3];
    const float* root1 = (const float*)d_in[4];
    const float* b1    = (const float*)d_in[5];
    const float* w2    = (const float*)d_in[6];
    const float* root2 = (const float*)d_in[7];
    const float* b2    = (const float*)d_in[8];
    float* out = (float*)d_out;

    // workspace layout (~80 MB)
    float* inv = (float*)d_ws;                              // N*R counts -> inverse counts
    float* y   = inv + (size_t)N_NODES * NREL;              // N*D relation-transformed feats
    float* h   = y   + (size_t)N_NODES * DIM;               // N*D layer-1 output
    unsigned short* wt1h = (unsigned short*)(h + (size_t)N_NODES * DIM);
    unsigned short* wt1l = wt1h + 9 * DIM * DIM;
    unsigned short* wt2h = wt1l + 9 * DIM * DIM;
    unsigned short* wt2l = wt2h + 9 * DIM * DIM;
    unsigned short* fhi  = wt2l + 9 * DIM * DIM;            // N*D bf16 hi of current feat
    unsigned short* flo  = fhi  + (size_t)N_NODES * DIM;    // N*D bf16 lo

    const int NR = N_NODES * NREL;
    const int B  = 256;
    const int gemm_blocks  = (N_NODES / 16) * 8 / 8;        // 3125 (8 waves/block)
    const int scat_blocks  = (NEDGE * 32) / B;              // 75000
    const int elem4_blocks = (N_NODES * DIM / 4) / B;       // 6250

    // --- shared preprocessing ---
    zero_f     <<<(NR + B - 1) / B, B, 0, stream>>>(inv, NR);
    count_edges<<<(NEDGE + B - 1) / B, B, 0, stream>>>(ei, et, inv);
    invert_cnt <<<(NR + B - 1) / B, B, 0, stream>>>(inv, NR);
    split_weights<<<(9 * DIM * DIM + B - 1) / B, B, 0, stream>>>(w1, root1, wt1h, wt1l);
    split_weights<<<(9 * DIM * DIM + B - 1) / B, B, 0, stream>>>(w2, root2, wt2h, wt2l);

    // --- layer 1: h = relu( sum_r scatter(x@w1_r) + x@root1 + b1 ) ---
    split_feat<<<elem4_blocks, B, 0, stream>>>(x, fhi, flo);
    gemm_bf16x3<<<gemm_blocks, B, 0, stream>>>(fhi, flo, wt1h + 8 * DIM * DIM,
                                               wt1l + 8 * DIM * DIM, b1, h);
    for (int r = 0; r < NREL; ++r) {
        gemm_bf16x3<<<gemm_blocks, B, 0, stream>>>(fhi, flo, wt1h + r * DIM * DIM,
                                                   wt1l + r * DIM * DIM, nullptr, y);
        scatter_rel<<<scat_blocks, B, 0, stream>>>(ei, et, y, inv, h, r);
    }
    relu_k<<<elem4_blocks, B, 0, stream>>>(h, N_NODES * DIM / 4);

    // --- layer 2: out = relu( sum_r scatter(h@w2_r) + h@root2 + b2 ) ---
    split_feat<<<elem4_blocks, B, 0, stream>>>(h, fhi, flo);
    gemm_bf16x3<<<gemm_blocks, B, 0, stream>>>(fhi, flo, wt2h + 8 * DIM * DIM,
                                               wt2l + 8 * DIM * DIM, b2, out);
    for (int r = 0; r < NREL; ++r) {
        gemm_bf16x3<<<gemm_blocks, B, 0, stream>>>(fhi, flo, wt2h + r * DIM * DIM,
                                                   wt2l + r * DIM * DIM, nullptr, y);
        scatter_rel<<<scat_blocks, B, 0, stream>>>(ei, et, y, inv, out, r);
    }
    relu_k<<<elem4_blocks, B, 0, stream>>>(out, N_NODES * DIM / 4);
}
